// Grouping_39694087750125
// MI455X (gfx1250) — matrix operand order repaired and checked by reading, not compiled, not run on hardware
//
#include <hip/hip_runtime.h>
#include <hip/hip_bf16.h>

// Problem constants (from the reference)
#define BB   4
#define NN   2048
#define DDIM 256
#define KK   16384
#define NTOP 5
#define EPSV 1e-6f

typedef __bf16 bf16;
typedef __attribute__((ext_vector_type(16))) __bf16 v16bf;
typedef __attribute__((ext_vector_type(8)))  __bf16 v8bf;
typedef __attribute__((ext_vector_type(8)))  float  v8f;
typedef __attribute__((ext_vector_type(4)))  unsigned int v4u;
typedef __attribute__((ext_vector_type(8)))  int    v8i;
typedef __attribute__((ext_vector_type(4)))  int    v4i;

// ---------------------------------------------------------------------------
// Zero the output buffer (poisoned by harness; atomics accumulate into it).
// ---------------------------------------------------------------------------
__global__ void zero_kernel(float* __restrict__ p, long long n4) {
    long long i = (long long)blockIdx.x * blockDim.x + threadIdx.x;
    long long stride = (long long)gridDim.x * blockDim.x;
    float4 z = {0.f, 0.f, 0.f, 0.f};
    for (; i < n4; i += stride) ((float4*)p)[i] = z;
}

// ---------------------------------------------------------------------------
// Convert a [rows x 256] f32 matrix to bf16 and compute per-row sum of squares.
// One 64-thread block per row, 4 elements per thread.
// ---------------------------------------------------------------------------
__global__ void prep_kernel(const float* __restrict__ src, bf16* __restrict__ dst,
                            float* __restrict__ sq) {
    const int row = blockIdx.x;
    const int tid = threadIdx.x;        // 0..63
    const float* p = src + (size_t)row * DDIM + tid * 4;
    float4 v = *(const float4*)p;
    bf16* q = dst + (size_t)row * DDIM + tid * 4;
    q[0] = (bf16)v.x; q[1] = (bf16)v.y; q[2] = (bf16)v.z; q[3] = (bf16)v.w;
    float s = v.x*v.x + v.y*v.y + v.z*v.z + v.w*v.w;
    __shared__ float red[64];
    red[tid] = s; __syncthreads();
    for (int o = 32; o > 0; o >>= 1) {
        if (tid < o) red[tid] += red[tid + o];
        __syncthreads();
    }
    if (tid == 0) sq[row] = red[0];
}

// 16-bit A fragment chunk (A 16x32 layout: per lane, elements 0..7 = K
// (half*8 + 0..7), elements 8..15 = K (16 + half*8 + 0..7)).
__device__ __forceinline__ v16bf load_fragA(const bf16* p /* row + c*32 + half*8 */) {
    v8bf lo = *(const v8bf*)p;
    v8bf hi = *(const v8bf*)(p + 16);
    v16bf f;
#pragma unroll
    for (int i = 0; i < 8; ++i) { f[i] = lo[i]; f[8 + i] = hi[i]; }
    return f;
}

// 16-bit B fragment chunk (B 32x16 layout: per lane, elements 0..7 = K
// (half*8 + 0..7), elements 8..15 = K (16 + half*8 + 0..7)),
// sourced from the LDS-staged centroid tile.
__device__ __forceinline__ v16bf load_fragB_lds(const bf16* p /* row + c*32 + half*8 */) {
    v8bf lo = *(const v8bf*)p;
    v8bf hi = *(const v8bf*)(p + 16);
    v16bf f;
#pragma unroll
    for (int i = 0; i < 8; ++i) { f[i] = lo[i]; f[8 + i] = hi[i]; }
    return f;
}

// ---------------------------------------------------------------------------
// TDM: DMA one 16-row x 512-byte centroid tile (bf16 [16][256]) into LDS.
// D# per CDNA5 ISA ch.8: group0 {count=1, lds_addr, global_addr, type=2},
// group1 {data_size=2B, tensor_dim0=256, tensor_dim1=K, tile 256x16,
// dim0_stride=256}; groups 2/3 (and the trailing group) zero — unused for
// 2-D tiles. 6-arg builtin form (clang-23 / therock-10.0 toolchain).
// ---------------------------------------------------------------------------
__device__ __forceinline__ void tdm_load_tile(const bf16* gsrc, unsigned int lds_off) {
    unsigned long long ga = (unsigned long long)(uintptr_t)gsrc;
    v4u g0;
    g0[0] = 1u;                                          // count=1 (valid user D#)
    g0[1] = lds_off;                                     // lds_addr (bytes)
    g0[2] = (unsigned int)(ga & 0xFFFFFFFFu);            // global_addr[31:0]
    g0[3] = (unsigned int)((ga >> 32) & 0x01FFFFFFu)     // global_addr[56:32]
          | 0x80000000u;                                 // type=2 ("image")
    v8i g1;
    g1[0] = 0x00010000;                                  // data_size=1 -> 2 bytes
    g1[1] = (int)(DDIM << 16);                           // tensor_dim0[15:0]=256
    g1[2] = (int)((KK & 0xFFFF) << 16);                  // tensor_dim1[15:0]
    g1[3] = (int)(DDIM << 16);                           // tile_dim0=256
    g1[4] = 16;                                          // tile_dim1=16
    g1[5] = DDIM;                                        // tensor_dim0_stride=256
    g1[6] = 0;
    g1[7] = 0;
    v4i gz4 = {0, 0, 0, 0};
    v8i gz8 = {0, 0, 0, 0, 0, 0, 0, 0};
    __builtin_amdgcn_tensor_load_to_lds(g0, g1, gz4, gz4, gz8, 0);
}

// ---------------------------------------------------------------------------
// Fused: cdist (WMMA bf16, f32 accum) -> per-row top-5 -> softmax ->
// scatter-add att*ass*(x-c), atomicMax(ass) weights.
// Block = 128 threads (4 waves) owning 64 feature rows (one 16-row M-tile per
// wave). All waves share one TDM-staged centroid k-tile per iteration
// (double-buffered in LDS), so each wave scans ALL K for its own rows:
// no cross-wave top-k merge, exact ascending-index tie-break.
// Top-5 selection streams s = c2[k] - 2*dot (order-equivalent to the
// distance: sqrt is monotonic and x2[row] is constant per row); the real
// distances are reconstructed only for the 5 winners.
// ---------------------------------------------------------------------------
__launch_bounds__(128)
__global__ void dist_topk_scatter_kernel(
    const float* __restrict__ feat,   // [B,N,D] f32
    const float* __restrict__ att,    // [B,N,1] f32
    const float* __restrict__ cent,   // [K,D]   f32
    const bf16*  __restrict__ featB,  // [B,N,D] bf16
    const bf16*  __restrict__ centB,  // [K,D]   bf16
    const float* __restrict__ x2g,    // [B,N]
    const float* __restrict__ c2g,    // [K]
    float* __restrict__ outG,         // [B,K,D] (accumulated via atomics)
    float* __restrict__ wout)         // [B,K]   (atomicMax of ass)
{
    const int tid  = threadIdx.x;
    const int wave = tid >> 5;          // 0..3 -> M-tile within block
    const int lane = tid & 31;
    const int half = lane >> 4;
    const int col  = lane & 15;

    const int b  = blockIdx.x / (NN / 64);
    const int n0 = (blockIdx.x % (NN / 64)) * 64;     // 64 rows per block
    const int nw = n0 + wave * 16;                    // this wave's 16 rows

    __shared__ __align__(16) bf16 bufB[2][16 * DDIM]; // 2 x 8KB staged k-tiles
    __shared__ float tileD[4][16][16];                // per-wave score tile
    __shared__ float rsS[4][16][NTOP];                // att*ass
    __shared__ int   riS[4][16][NTOP];                // centroid idx

    // Kick off TDM for k-tile 0 while we load A fragments.
    if (wave == 0)
        tdm_load_tile(centB, (unsigned int)(uintptr_t)(void*)&bufB[0][0]);

    // A fragments: this wave's 16 feature rows, resident in VGPRs throughout.
    const bf16* arow = featB + ((size_t)b * NN + nw + col) * DDIM;
    v16bf afrag[8];
#pragma unroll
    for (int c = 0; c < 8; ++c) afrag[c] = load_fragA(arow + c * 32 + half * 8);

    // Private top-5 scores (ascending) for row M = lane (lanes 0..15 only).
    float d0 = 3.4e38f, d1 = 3.4e38f, d2 = 3.4e38f, d3 = 3.4e38f, d4 = 3.4e38f;
    int   i0 = 0, i1 = 0, i2 = 0, i3 = 0, i4 = 0;

    if (wave == 0) __builtin_amdgcn_s_wait_tensorcnt(0);
    __syncthreads();                                   // tile 0 staged

    const int TILES = KK / 16;                         // 1024 shared k-tiles
    for (int t = 0; t < TILES; ++t) {
        const int cur = t & 1;
        // Prefetch next k-tile via TDM into the spare buffer. The barrier at
        // the end of iteration t-1 guaranteed all reads of that buffer done.
        if (wave == 0 && (t + 1) < TILES)
            tdm_load_tile(centB + (size_t)(t + 1) * 16 * DDIM,
                          (unsigned int)(uintptr_t)(void*)&bufB[1 - cur][0]);

        const int kg = t * 16 + col;
        const bf16* brow = &bufB[cur][(size_t)col * DDIM];

        // Load the whole B tile for this wave's column first (clustered
        // ds_load_b128s), then run the WMMA chain back-to-back.
        v16bf bfrag[8];
#pragma unroll
        for (int c = 0; c < 8; ++c)
            bfrag[c] = load_fragB_lds(brow + c * 32 + half * 8);

        v8f acc = {0.f, 0.f, 0.f, 0.f, 0.f, 0.f, 0.f, 0.f};
#pragma unroll
        for (int c = 0; c < 8; ++c)
            acc = __builtin_amdgcn_wmma_f32_16x16x32_bf16(
                false, afrag[c], false, bfrag[c], (short)0, acc, false, false);

        // Selection score: s = c2[k] - 2*dot  (no sqrt, no x2 needed for
        // ordering within a row).
        const float c2v = c2g[kg];
#pragma unroll
        for (int i = 0; i < 8; ++i)
            tileD[wave][half * 8 + i][col] = fmaf(-2.0f, acc[i], c2v);

        // Wave-private LDS region; DS ops within a wave retire in order.
        if (lane < 16) {
            const int kb = t * 16;
#pragma unroll
            for (int j = 0; j < 16; ++j) {
                float dv = tileD[wave][lane][j];
                if (dv < d4) {
                    int kv = kb + j;   // k ascends -> strict '<' keeps the
                    if (dv < d3) {     // reference's smaller-index tiebreak
                        d4 = d3; i4 = i3;
                        if (dv < d2) {
                            d3 = d2; i3 = i2;
                            if (dv < d1) {
                                d2 = d1; i2 = i1;
                                if (dv < d0) { d1 = d0; i1 = i0; d0 = dv; i0 = kv; }
                                else         { d1 = dv; i1 = kv; }
                            } else { d2 = dv; i2 = kv; }
                        } else { d3 = dv; i3 = kv; }
                    } else { d4 = dv; i4 = kv; }
                }
            }
        }

        if (wave == 0) __builtin_amdgcn_s_wait_tensorcnt(0);
        __syncthreads();   // next tile staged; all reads of bufB[cur] done
    }

    // Reconstruct the 5 winning distances, softmax (base 1.0), weights.
    if (lane < 16) {
        const int r = lane;
        const float x2r = x2g[b * NN + nw + r];
        float t0 = sqrtf(fmaxf(x2r + d0, 0.f));
        float t1 = sqrtf(fmaxf(x2r + d1, 0.f));
        float t2 = sqrtf(fmaxf(x2r + d2, 0.f));
        float t3 = sqrtf(fmaxf(x2r + d3, 0.f));
        float t4 = sqrtf(fmaxf(x2r + d4, 0.f));
        float e0 = 1.0f;                  // expf(t0 - t0)
        float e1 = expf(t0 - t1);
        float e2 = expf(t0 - t2);
        float e3 = expf(t0 - t3);
        float e4 = expf(t0 - t4);
        float ssum = e0 + e1 + e2 + e3 + e4;
        const float av = att[(size_t)b * NN + nw + r];
        float as0 = e0 / ssum, as1 = e1 / ssum, as2 = e2 / ssum,
              as3 = e3 / ssum, as4 = e4 / ssum;
        rsS[wave][r][0] = av * as0; riS[wave][r][0] = i0;
        rsS[wave][r][1] = av * as1; riS[wave][r][1] = i1;
        rsS[wave][r][2] = av * as2; riS[wave][r][2] = i2;
        rsS[wave][r][3] = av * as3; riS[wave][r][3] = i3;
        rsS[wave][r][4] = av * as4; riS[wave][r][4] = i4;
        // non-negative floats: uint ordering == float ordering
        float* wb = wout + (size_t)b * KK;
        atomicMax((unsigned int*)&wb[i0], __float_as_uint(as0));
        atomicMax((unsigned int*)&wb[i1], __float_as_uint(as1));
        atomicMax((unsigned int*)&wb[i2], __float_as_uint(as2));
        atomicMax((unsigned int*)&wb[i3], __float_as_uint(as3));
        atomicMax((unsigned int*)&wb[i4], __float_as_uint(as4));
    }
    __syncthreads();

    // Scatter-add descriptors: each wave covers its 16 rows x 5 sel x 256 dims.
    for (int r = 0; r < 16; ++r) {
        const float* xrow = feat + ((size_t)b * NN + nw + r) * DDIM;
#pragma unroll
        for (int j = 0; j < NTOP; ++j) {
            const int k = riS[wave][r][j];
            const float sc = rsS[wave][r][j];
            const float* crow = cent + (size_t)k * DDIM;
            float* orow = outG + ((size_t)b * KK + k) * DDIM;
            for (int d = lane; d < DDIM; d += 32)
                atomicAdd(&orow[d], sc * (xrow[d] - crow[d]));
        }
    }
}

// ---------------------------------------------------------------------------
// In-place L2 normalize each [256] row: row /= (||row|| + eps).
// One 64-thread block per (b,k) row; float4 per thread.
// ---------------------------------------------------------------------------
__global__ void normalize_kernel(float* __restrict__ outG) {
    const int row = blockIdx.x;         // 0 .. B*K-1
    const int tid = threadIdx.x;        // 0..63
    float4* p = (float4*)(outG + (size_t)row * DDIM) + tid;
    float4 v = *p;
    float s = v.x*v.x + v.y*v.y + v.z*v.z + v.w*v.w;
    __shared__ float red[64];
    red[tid] = s; __syncthreads();
    for (int o = 32; o > 0; o >>= 1) {
        if (tid < o) red[tid] += red[tid + o];
        __syncthreads();
    }
    const float inv = 1.0f / (sqrtf(red[0]) + EPSV);
    v.x *= inv; v.y *= inv; v.z *= inv; v.w *= inv;
    *p = v;
}

// ---------------------------------------------------------------------------
extern "C" void kernel_launch(void* const* d_in, const int* in_sizes, int n_in,
                              void* d_out, int out_size, void* d_ws, size_t ws_size,
                              hipStream_t stream) {
    const float* feat = (const float*)d_in[0];   // [B,N,D]
    const float* att  = (const float*)d_in[1];   // [B,N,1]
    const float* cent = (const float*)d_in[2];   // [K,D]

    float* outG = (float*)d_out;                       // [B,K,D]
    float* wout = outG + (size_t)BB * KK * DDIM;       // [B,K]

    // Workspace layout (~12.1 MB)
    char* ws = (char*)d_ws;
    bf16* featB = (bf16*)ws;
    size_t off = (size_t)BB * NN * DDIM * sizeof(bf16);
    bf16* centB = (bf16*)(ws + off);
    off += (size_t)KK * DDIM * sizeof(bf16);
    float* x2 = (float*)(ws + off);
    off += (size_t)BB * NN * sizeof(float);
    float* c2 = (float*)(ws + off);

    zero_kernel<<<2048, 256, 0, stream>>>(outG, (long long)out_size / 4);
    prep_kernel<<<KK, 64, 0, stream>>>(cent, centB, c2);
    prep_kernel<<<BB * NN, 64, 0, stream>>>(feat, featB, x2);
    dist_topk_scatter_kernel<<<BB * (NN / 64), 128, 0, stream>>>(
        feat, att, cent, featB, centB, x2, c2, outG, wout);
    normalize_kernel<<<BB * KK, 64, 0, stream>>>(outG);
}
